// RNNModel_46712064312127
// MI455X (gfx1250) — matrix operand order repaired and checked by reading, not compiled, not run on hardware
//
#include <hip/hip_runtime.h>

// ---------------------------------------------------------------------------
// Model dims (from reference)
#define T_LEN 512
#define B_SZ  32
#define E_DIM 300
#define E_PAD 320      // pad K to multiple of 32 for WMMA
#define H0_   256
#define H1_   128

typedef unsigned short ushort_t;
typedef __attribute__((ext_vector_type(16))) __bf16 v16bf;
typedef __attribute__((ext_vector_type(8)))  float  v8f;

union Frag16 { uint4  u[2]; v16bf v; };   // 32 bytes/lane: one 16-bit WMMA A/B fragment
union Frag32 { float4 f4[2]; v8f  v; };   // 32 bytes/lane: one f32 C/D fragment

__device__ __forceinline__ unsigned short f2bf(float f) {
    union { float f; unsigned u; } x; x.f = f;
    unsigned r = x.u + 0x7FFFu + ((x.u >> 16) & 1u);   // RNE
    return (unsigned short)(r >> 16);
}
__device__ __forceinline__ float bf2f(unsigned short h) {
    union { unsigned u; float f; } x; x.u = ((unsigned)h) << 16;
    return x.f;
}
__device__ __forceinline__ float sigf(float x) { return 1.f / (1.f + __expf(-x)); }

// ---------------------------------------------------------------------------
// Pack a (possibly concatenated) f32 weight [K,N] row-major into bf16 WMMA
// B-fragment tiles. Tile (nt,kt) id = nt*KT + kt, 512 bf16 = 1KB each.
// Element (lane,j): K = kt*32 + (lane>=16 ? 8 : 0) + (j<8 ? j : j+8) ; N = nt*16 + (lane&15).
__global__ void pack_b_kernel(const float* __restrict__ srcA,
                              const float* __restrict__ srcB, int Nsplit,
                              int Ktot, int Ntot, int Kvalid,
                              ushort_t* __restrict__ dst) {
    int KT = Ktot >> 5, NT = Ntot >> 4;
    int idx = blockIdx.x * blockDim.x + threadIdx.x;
    if (idx >= NT * KT * 512) return;
    int e = idx & 511, tile = idx >> 9;
    int kt = tile % KT, nt = tile / KT;
    int lane = e >> 4, j = e & 15;
    int k = kt * 32 + ((lane >= 16) ? 8 : 0) + ((j < 8) ? j : j + 8);
    int n = nt * 16 + (lane & 15);
    float v = 0.f;
    if (k < Kvalid) {
        if (n < Nsplit) v = srcA[(size_t)k * Nsplit + n];
        else            v = srcB[(size_t)k * (Ntot - Nsplit) + (n - Nsplit)];
    }
    dst[idx] = f2bf(v);
}

__global__ void concat2_f32(const float* __restrict__ a, int na,
                            const float* __restrict__ b, int nb,
                            float* __restrict__ dst) {
    int i = blockIdx.x * blockDim.x + threadIdx.x;
    if (i < na) dst[i] = a[i];
    else if (i < na + nb) dst[i] = b[i - na];
}

// Embedding gather -> time-major bf16 [T, B, E_PAD] (zero padded K)
__global__ void embed_kernel(const int* __restrict__ tokens,
                             const float* __restrict__ emb,
                             ushort_t* __restrict__ x) {
    int idx = blockIdx.x * blockDim.x + threadIdx.x;
    if (idx >= T_LEN * B_SZ * E_PAD) return;
    int k = idx % E_PAD;
    int row = idx / E_PAD;            // t*B + b
    int b = row % B_SZ, t = row / B_SZ;
    float v = 0.f;
    if (k < E_DIM) {
        int tok = tokens[b * T_LEN + t];
        v = emb[(size_t)tok * E_DIM + k];
    }
    x[idx] = f2bf(v);
}

// ---------------------------------------------------------------------------
// Generic bf16 WMMA GEMM:  A [MT*16, K] bf16 row-major  x  Bpk (packed tiles)
// One wave computes a 16 x 64 output strip (1 M-tile x 4 N-tiles).
// Ping-pong double-buffered k-loop (KT must be even -- true for all K here):
// loads for k+2 are issued while k+1's WMMAs run, so the compiler emits
// partial s_wait_loadcnt and no fragment register copies.
// out_mode 0: raw f32 C-fragment tiles (tile id = mt*NT + nt, 1KB each)
// out_mode 1: bf16 row-major [MT*16, NT*16] with optional bias+relu
__global__ void __launch_bounds__(256)
gemm_wmma(const ushort_t* __restrict__ A, int K,
          const ushort_t* __restrict__ Bpk, int MT, int NT,
          const float* __restrict__ bias, int relu, int out_mode,
          float* __restrict__ outp, ushort_t* __restrict__ outb) {
    int KT = K >> 5;
    int wid  = (blockIdx.x * blockDim.x + threadIdx.x) >> 5;
    int lane = threadIdx.x & 31;
    int NG = NT >> 2;
    if (wid >= MT * NG) return;
    int mt = wid / NG, nt0 = (wid % NG) * 4;

    v8f acc[4] = {};
    const int row  = mt * 16 + (lane & 15);
    const int aofs = (lane >= 16) ? 8 : 0;      // A-matrix lane K offset (ISA layout)
    const ushort_t* apBase = A + (size_t)row * K + aofs;

    auto loadA = [&](Frag16& a, int kt) {
        const ushort_t* ap = apBase + kt * 32;
        a.u[0] = *(const uint4*)(ap);
        a.u[1] = *(const uint4*)(ap + 16);
    };
    auto loadB = [&](Frag16* b, int kt) {
        #pragma unroll
        for (int q = 0; q < 4; ++q) {
            const ushort_t* bp =
                Bpk + ((size_t)(nt0 + q) * KT + kt) * 512 + lane * 16;
            b[q].u[0] = *(const uint4*)(bp);
            b[q].u[1] = *(const uint4*)(bp + 8);
        }
    };

    Frag16 a0, a1, b0[4], b1[4];
    loadA(a0, 0); loadB(b0, 0);
    loadA(a1, 1); loadB(b1, 1);
    for (int kt = 0; kt < KT; kt += 2) {
        #pragma unroll
        for (int q = 0; q < 4; ++q)
            acc[q] = __builtin_amdgcn_wmma_f32_16x16x32_bf16(
                false, a0.v, false, b0[q].v, (short)0, acc[q], false, false);
        if (kt + 2 < KT) { loadA(a0, kt + 2); loadB(b0, kt + 2); }
        #pragma unroll
        for (int q = 0; q < 4; ++q)
            acc[q] = __builtin_amdgcn_wmma_f32_16x16x32_bf16(
                false, a1.v, false, b1[q].v, (short)0, acc[q], false, false);
        if (kt + 3 < KT) { loadA(a1, kt + 3); loadB(b1, kt + 3); }
    }

    const int Nout = NT * 16;
    #pragma unroll
    for (int q = 0; q < 4; ++q) {
        int n = (nt0 + q) * 16 + (lane & 15);
        float bv = bias ? bias[n] : 0.f;
        v8f r = acc[q];
        #pragma unroll
        for (int i = 0; i < 8; ++i) {
            float v = r[i] + bv;
            if (relu) v = fmaxf(v, 0.f);
            r[i] = v;
        }
        if (out_mode == 0) {
            Frag32 f; f.v = r;
            float* dp = outp + ((size_t)mt * NT + nt0 + q) * 256 + lane * 8;
            ((float4*)dp)[0] = f.f4[0];
            ((float4*)dp)[1] = f.f4[1];
        } else {
            int rbase = mt * 16 + ((lane >= 16) ? 8 : 0);
            #pragma unroll
            for (int i = 0; i < 8; ++i)
                outb[(size_t)(rbase + i) * Nout + n] = f2bf(r[i]);
        }
    }
}

// ---------------------------------------------------------------------------
// Persistent LSTM recurrence. One block per direction; wave w owns hidden
// columns [16w,16w+16) and ALL FOUR gate blocks for them, so c/h update is
// wave-local in accumulator layout. h broadcast via LDS each step.
// xg: precomputed input gates in raw C-fragment layout (f32), tile id =
// (t*2+m) * (2*4H/16) + d*(4H/16) + q*(H/16) + w.
// Ping-pong double-buffered U fragments; next step's xg tiles prefetched.
template <int H>
__global__ void __launch_bounds__(512)
lstm_rec(const float* __restrict__ xg,
         const ushort_t* __restrict__ UpkF, const ushort_t* __restrict__ UpkB,
         ushort_t* __restrict__ hout, int outStride) {
    constexpr int NW = H / 16;           // waves per block
    constexpr int KT = H / 32;           // even for H=128/256
    constexpr int NT_TOT = 2 * (4 * H / 16);
    const int d    = blockIdx.x;         // 0=fwd, 1=bwd
    const ushort_t* Upk = d ? UpkB : UpkF;
    const int lane = threadIdx.x & 31;
    const int w    = threadIdx.x >> 5;

    __shared__ ushort_t hsh[32][H + 8];  // +8 bf16 pad: kill LDS bank conflicts
    for (int i = threadIdx.x; i < 32 * (H + 8); i += blockDim.x)
        (&hsh[0][0])[i] = 0;
    __syncthreads();

    v8f c[2] = {};                       // cell state, lives in C/D layout
    const int aofs = (lane >= 16) ? 8 : 0;
    const int arow = lane & 15;
    const int col  = w * 16 + (lane & 15);
    const int brow = (lane >= 16) ? 8 : 0;
    const int ntq  = d * (4 * H / 16) + w;   // + q*(H/16)

    auto loadU = [&](Frag16* b, int kt) {
        #pragma unroll
        for (int q = 0; q < 4; ++q) {
            const ushort_t* bp =
                Upk + ((size_t)(q * NW + w) * KT + kt) * 512 + lane * 16;
            b[q].u[0] = *(const uint4*)(bp);
            b[q].u[1] = *(const uint4*)(bp + 8);
        }
    };
    auto loadH = [&](Frag16& a0, Frag16& a1, int kt) {
        const ushort_t* ap0 = &hsh[arow][kt * 32 + aofs];
        a0.u[0] = *(const uint4*)(ap0);  a0.u[1] = *(const uint4*)(ap0 + 16);
        const ushort_t* ap1 = &hsh[16 + arow][kt * 32 + aofs];
        a1.u[0] = *(const uint4*)(ap1);  a1.u[1] = *(const uint4*)(ap1 + 16);
    };

    for (int s = 0; s < T_LEN; ++s) {
        const int t = d ? (T_LEN - 1 - s) : s;

        // prefetch next step's xg fragment tiles (global_prefetch_b8):
        // shortens the serial chain's accumulator-init loads
        if (s + 1 < T_LEN) {
            int tn = d ? (T_LEN - 2 - s) : (s + 1);
            #pragma unroll
            for (int m = 0; m < 2; ++m) {
                #pragma unroll
                for (int q = 0; q < 4; ++q) {
                    const float* pp = xg +
                        ((size_t)(tn * 2 + m) * NT_TOT + ntq + q * (H / 16)) * 256 +
                        lane * 8;
                    __builtin_prefetch(pp, 0, 3);
                }
            }
        }

        // init accumulators from precomputed x@W + bias (raw fragment loads)
        v8f acc[2][4];
        #pragma unroll
        for (int m = 0; m < 2; ++m) {
            int mt = t * 2 + m;
            #pragma unroll
            for (int q = 0; q < 4; ++q) {
                const float* dp = xg +
                    ((size_t)mt * NT_TOT + ntq + q * (H / 16)) * 256 + lane * 8;
                Frag32 f;
                f.f4[0] = ((const float4*)dp)[0];
                f.f4[1] = ((const float4*)dp)[1];
                acc[m][q] = f.v;
            }
        }

        // h_{t-1} @ U  (A from LDS, B from L2-resident packed U), ping-pong
        Frag16 b0[4], b1[4];
        loadU(b0, 0);
        loadU(b1, 1);
        #pragma unroll
        for (int kt = 0; kt < KT; kt += 2) {
            Frag16 a0, a1;
            loadH(a0, a1, kt);
            #pragma unroll
            for (int q = 0; q < 4; ++q) {
                acc[0][q] = __builtin_amdgcn_wmma_f32_16x16x32_bf16(
                    false, a0.v, false, b0[q].v, (short)0, acc[0][q], false, false);
                acc[1][q] = __builtin_amdgcn_wmma_f32_16x16x32_bf16(
                    false, a1.v, false, b0[q].v, (short)0, acc[1][q], false, false);
            }
            if (kt + 2 < KT) loadU(b0, kt + 2);
            Frag16 a2, a3;
            loadH(a2, a3, kt + 1);
            #pragma unroll
            for (int q = 0; q < 4; ++q) {
                acc[0][q] = __builtin_amdgcn_wmma_f32_16x16x32_bf16(
                    false, a2.v, false, b1[q].v, (short)0, acc[0][q], false, false);
                acc[1][q] = __builtin_amdgcn_wmma_f32_16x16x32_bf16(
                    false, a3.v, false, b1[q].v, (short)0, acc[1][q], false, false);
            }
            if (kt + 3 < KT) loadU(b1, kt + 3);
        }
        __syncthreads();   // all waves done reading hsh

        // gates + state update, entirely in accumulator layout
        #pragma unroll
        for (int m = 0; m < 2; ++m) {
            #pragma unroll
            for (int i = 0; i < 8; ++i) {
                float iv = sigf(acc[m][0][i]);
                float fv = sigf(acc[m][1][i]);
                float gv = tanhf(acc[m][2][i]);
                float ov = sigf(acc[m][3][i]);
                float cv = fv * c[m][i] + iv * gv;
                c[m][i] = cv;
                float hv = ov * tanhf(cv);
                unsigned short hb = f2bf(hv);
                int b = m * 16 + brow + i;
                hsh[b][col] = hb;
                hout[((size_t)t * B_SZ + b) * outStride + d * H + col] = hb;
            }
        }
        __syncthreads();   // hsh updated for next step
    }
}

// ---------------------------------------------------------------------------
// Classifier head: logits = m2 @ clfW + clfb ; softmax. One thread per (t,b).
__global__ void head_kernel(const ushort_t* __restrict__ m2,
                            const float* __restrict__ clfW,
                            const float* __restrict__ clfb,
                            float* __restrict__ out) {
    int r = blockIdx.x * blockDim.x + threadIdx.x;    // t*32 + b
    if (r >= T_LEN * B_SZ) return;
    int b = r % B_SZ, t = r / B_SZ;
    float lg[3];
    #pragma unroll
    for (int cc = 0; cc < 3; ++cc) {
        float s = clfb[cc];
        for (int k = 0; k < 64; ++k)
            s += bf2f(m2[(size_t)r * 64 + k]) * clfW[k * 3 + cc];
        lg[cc] = s;
    }
    float mx = fmaxf(lg[0], fmaxf(lg[1], lg[2]));
    float e0 = __expf(lg[0] - mx), e1 = __expf(lg[1] - mx), e2 = __expf(lg[2] - mx);
    float inv = 1.f / (e0 + e1 + e2);
    float* op = out + ((size_t)b * T_LEN + t) * 3;
    op[0] = e0 * inv; op[1] = e1 * inv; op[2] = e2 * inv;
}

// ---------------------------------------------------------------------------
extern "C" void kernel_launch(void* const* d_in, const int* in_sizes, int n_in,
                              void* d_out, int out_size, void* d_ws, size_t ws_size,
                              hipStream_t stream) {
    const int*   tokens = (const int*)  d_in[0];
    const float* emb    = (const float*)d_in[1];
    const float* W0f = (const float*)d_in[2];
    const float* U0f = (const float*)d_in[3];
    const float* b0f = (const float*)d_in[4];
    const float* W0b = (const float*)d_in[5];
    const float* U0b = (const float*)d_in[6];
    const float* b0b = (const float*)d_in[7];
    const float* W1f = (const float*)d_in[8];
    const float* U1f = (const float*)d_in[9];
    const float* b1f = (const float*)d_in[10];
    const float* W1b = (const float*)d_in[11];
    const float* U1b = (const float*)d_in[12];
    const float* b1b = (const float*)d_in[13];
    const float* mW1 = (const float*)d_in[14];
    const float* mb1 = (const float*)d_in[15];
    const float* mW2 = (const float*)d_in[16];
    const float* mb2 = (const float*)d_in[17];
    const float* cW  = (const float*)d_in[18];
    const float* cb  = (const float*)d_in[19];
    float* out = (float*)d_out;

    char* p = (char*)d_ws;
    auto carve = [&](size_t bytes) {
        char* r = p; p += (bytes + 255) & ~(size_t)255; return r;
    };
    ushort_t* x0   = (ushort_t*)carve((size_t)16384 * E_PAD * 2);   // [T,B,320] bf16
    float*    xg0  = (float*)   carve((size_t)1024 * 128 * 1024);   // frag tiles f32
    ushort_t* h0   = (ushort_t*)carve((size_t)16384 * 512 * 2);     // [T,B,512] bf16
    float*    xg1  = (float*)   carve((size_t)1024 * 64 * 1024);
    ushort_t* h1   = (ushort_t*)carve((size_t)16384 * 256 * 2);
    ushort_t* m1   = (ushort_t*)carve((size_t)16384 * 128 * 2);
    ushort_t* m2b  = (ushort_t*)carve((size_t)16384 * 64 * 2);
    ushort_t* Bpk0 = (ushort_t*)carve((size_t)128 * 10 * 1024);     // [320,2048]
    ushort_t* Up0f = (ushort_t*)carve((size_t)64 * 8 * 1024);       // [256,1024]
    ushort_t* Up0b = (ushort_t*)carve((size_t)64 * 8 * 1024);
    ushort_t* Bpk1 = (ushort_t*)carve((size_t)64 * 16 * 1024);      // [512,1024]
    ushort_t* Up1f = (ushort_t*)carve((size_t)32 * 4 * 1024);       // [128,512]
    ushort_t* Up1b = (ushort_t*)carve((size_t)32 * 4 * 1024);
    ushort_t* Wp1  = (ushort_t*)carve((size_t)8 * 8 * 1024);        // [256,128]
    ushort_t* Wp2  = (ushort_t*)carve((size_t)4 * 4 * 1024);        // [128,64]
    float*    bc0  = (float*)carve(2048 * 4);
    float*    bc1  = (float*)carve(1024 * 4);

    auto cdiv = [](int a, int b) { return (a + b - 1) / b; };

    // ---- one-time (per call) weight repack into WMMA fragment layout ----
    pack_b_kernel<<<cdiv(128*10*512,256),256,0,stream>>>(W0f, W0b, 1024, 320, 2048, 300, Bpk0);
    pack_b_kernel<<<cdiv(64*8*512, 256),256,0,stream>>>(U0f, U0f, 1024, 256, 1024, 256, Up0f);
    pack_b_kernel<<<cdiv(64*8*512, 256),256,0,stream>>>(U0b, U0b, 1024, 256, 1024, 256, Up0b);
    pack_b_kernel<<<cdiv(64*16*512,256),256,0,stream>>>(W1f, W1b,  512, 512, 1024, 512, Bpk1);
    pack_b_kernel<<<cdiv(32*4*512, 256),256,0,stream>>>(U1f, U1f,  512, 128,  512, 128, Up1f);
    pack_b_kernel<<<cdiv(32*4*512, 256),256,0,stream>>>(U1b, U1b,  512, 128,  512, 128, Up1b);
    pack_b_kernel<<<cdiv(8*8*512,  256),256,0,stream>>>(mW1, mW1,  128, 256,  128, 256, Wp1);
    pack_b_kernel<<<cdiv(4*4*512,  256),256,0,stream>>>(mW2, mW2,   64, 128,   64, 128, Wp2);
    concat2_f32<<<cdiv(2048,256),256,0,stream>>>(b0f, 1024, b0b, 1024, bc0);
    concat2_f32<<<cdiv(1024,256),256,0,stream>>>(b1f,  512, b1b,  512, bc1);

    // ---- embedding gather, time-major ----
    embed_kernel<<<cdiv(16384*E_PAD,256),256,0,stream>>>(tokens, emb, x0);

    // ---- layer 0: input gates (parallel WMMA), then serial recurrence ----
    gemm_wmma<<<cdiv(1024*(128/4)*32,256),256,0,stream>>>(
        x0, 320, Bpk0, 1024, 128, bc0, 0, 0, xg0, nullptr);
    lstm_rec<H0_><<<2, 512, 0, stream>>>(xg0, Up0f, Up0b, h0, 2 * H0_);

    // ---- layer 1 ----
    gemm_wmma<<<cdiv(1024*(64/4)*32,256),256,0,stream>>>(
        h0, 512, Bpk1, 1024, 64, bc1, 0, 0, xg1, nullptr);
    lstm_rec<H1_><<<2, 256, 0, stream>>>(xg1, Up1f, Up1b, h1, 2 * H1_);

    // ---- MLP (fused bias+relu, WMMA) ----
    gemm_wmma<<<cdiv(1024*(8/4)*32,256),256,0,stream>>>(
        h1, 256, Wp1, 1024, 8, mb1, 1, 1, nullptr, m1);
    gemm_wmma<<<cdiv(1024*(4/4)*32,256),256,0,stream>>>(
        m1, 128, Wp2, 1024, 4, mb2, 1, 1, nullptr, m2b);

    // ---- classifier + softmax ----
    head_kernel<<<cdiv(16384,256),256,0,stream>>>(m2b, cW, cb, out);
}
